// FlashMHA_67207648247915
// MI455X (gfx1250) — compile-verified, matrix operand order checked by reading, not run on hardware
//
#include <hip/hip_runtime.h>
#include <hip/hip_bf16.h>

// ---------------------------------------------------------------------------
// FlashMHA forward for MI455X (gfx1250), bf16 WMMA with f32 accumulation.
// Z=2, N=2048, Dm=1024, H=16, Dk=64.
// Attention K/V tiles staged into LDS via GLOBAL_LOAD_ASYNC_TO_LDS_B128
// (ASYNCcnt-tracked, double-buffered), shared by all 8 waves of a block.
// ---------------------------------------------------------------------------

typedef __bf16 bhalf;
typedef __attribute__((ext_vector_type(16))) __bf16 v16bf;
typedef __attribute__((ext_vector_type(8)))  __bf16 v8bf;
typedef __attribute__((ext_vector_type(8)))  float  v8f;

#define DMODEL 1024
#define NSEQ   2048
#define NHEADS 16
#define DKH    64
#define ZB     2
#define MROWS  (ZB * NSEQ)   // 4096 fused (z,n) rows

// ---- CDNA5 async-copy plumbing --------------------------------------------
#if defined(__AMDGCN__) && __has_builtin(__builtin_amdgcn_global_load_async_to_lds_b128)
#define HAVE_ASYNC_LDS 1
#pragma message("CDNA5 probe: __builtin_amdgcn_global_load_async_to_lds_b128 = YES")
#else
#define HAVE_ASYNC_LDS 0
#pragma message("CDNA5 probe: __builtin_amdgcn_global_load_async_to_lds_b128 = NO")
#endif
#if defined(__AMDGCN__) && __has_builtin(__builtin_amdgcn_s_wait_asynccnt)
#define HAVE_WAIT_ASYNC 1
#else
#define HAVE_WAIT_ASYNC 0
#endif

// Exact parameter types per compiler diagnostic: v4i in AS1 (global) and
// AS3 (LDS).  Built via integer casts (inttoptr); truncating a generic LDS
// address to 32 bits yields the LDS offset per the ISA aperture mapping.
typedef int v4i __attribute__((vector_size(4 * sizeof(int))));
typedef __attribute__((address_space(1))) v4i* as1_v4i;
typedef __attribute__((address_space(3))) v4i* as3_v4i;

// 16-byte global -> LDS copy (async on CDNA5, sync fallback otherwise).
__device__ __forceinline__ void async_cp16(const bhalf* g, bhalf* l) {
#if HAVE_ASYNC_LDS
  __builtin_amdgcn_global_load_async_to_lds_b128(
      (as1_v4i)(unsigned long long)(uintptr_t)g,
      (as3_v4i)(unsigned int)(uintptr_t)l,
      0, 0);
#else
  *(v8bf*)l = *(const v8bf*)g;
#endif
}

__device__ __forceinline__ void wait_async0() {
#if HAVE_WAIT_ASYNC
  __builtin_amdgcn_s_wait_asynccnt(0);
#elif HAVE_ASYNC_LDS && defined(__AMDGCN__)
  asm volatile("s_wait_asynccnt 0" ::: "memory");
#endif
}

// ---- WMMA fragment helpers -------------------------------------------------
// 16-bit A/B fragment (16 x 32 contraction slice), per-lane element order per
// CDNA5 ISA: element i -> K = (i<8?0:16) + 8*half + (i&7).  Caller passes a
// pointer already offset by (row * stride + kchunk_base + 8*half); we load two
// 16-byte chunks 32 bytes apart.
__device__ __forceinline__ v16bf ld_frag16(const bhalf* p) {
  v8bf lo = *(const v8bf*)(p);        // K = kb .. kb+7
  v8bf hi = *(const v8bf*)(p + 16);   // K = kb+16 .. kb+23
  v16bf r;
#pragma unroll
  for (int i = 0; i < 8; ++i) { r[i] = lo[i]; r[i + 8] = hi[i]; }
  return r;
}

__device__ __forceinline__ v8f wmma_bf16(v16bf a, v16bf b, v8f c) {
  return __builtin_amdgcn_wmma_f32_16x16x32_bf16(false, a, false, b,
                                                 (short)0, c, false, false);
}

// ---- fp32 -> bf16 converters ----------------------------------------------
__global__ void k_cvt(const float* __restrict__ in, bhalf* __restrict__ out, int n) {
  int i = blockIdx.x * 256 + threadIdx.x;
  if (i < n) out[i] = (bhalf)in[i];
}

// proj [H][Dm][Dk] fp32 -> wt [H][Dk][Dm] bf16 (transposed so WMMA B-fragments
// are contiguous per lane).
__global__ void k_cvt_wT(const float* __restrict__ in, bhalf* __restrict__ out) {
  int i = blockIdx.x * 256 + threadIdx.x;   // H*Dm*Dk = 1,048,576
  int h = i / (DMODEL * DKH);
  int r = i % (DMODEL * DKH);
  int d = r / DKH, k = r % DKH;
  out[((size_t)h * DKH + k) * DMODEL + d] = (bhalf)in[i];
}

// ---- QKV projection:  [4096 x 1024] x [1024 x 64] per head ----------------
// grid (32, H, 3).  8 waves/block; wave computes a 16x64 tile.
// which==0 -> Q [z,h,n,dk], which==1 -> K [z,h,n,dk], which==2 -> V^T [z,h,dk,n]
__global__ void __launch_bounds__(256)
k_qkv(const bhalf* __restrict__ xb,
      const bhalf* __restrict__ wq, const bhalf* __restrict__ wk, const bhalf* __restrict__ wv,
      const float* __restrict__ bq, const float* __restrict__ bk, const float* __restrict__ bv,
      bhalf* __restrict__ qo, bhalf* __restrict__ ko, bhalf* __restrict__ vo) {
  const int lane = threadIdx.x & 31;
  const int wave = threadIdx.x >> 5;
  const int half = lane >> 4;
  const int l16  = lane & 15;
  const int h     = blockIdx.y;
  const int which = blockIdx.z;
  const bhalf* w    = (which == 0) ? wq : (which == 1) ? wk : wv;
  const float* bias = (which == 0) ? bq : (which == 1) ? bk : bv;
  const int mbase = blockIdx.x * 128 + wave * 16;

  const v8f vzero = {0.f, 0.f, 0.f, 0.f, 0.f, 0.f, 0.f, 0.f};
  v8f acc[4];
#pragma unroll
  for (int t = 0; t < 4; ++t) acc[t] = vzero;

  const bhalf* arow = xb + (size_t)(mbase + l16) * DMODEL + half * 8;
  const bhalf* wh   = w + (size_t)h * DKH * DMODEL;

  for (int kc = 0; kc < DMODEL / 32; ++kc) {
    v16bf a = ld_frag16(arow + kc * 32);
#pragma unroll
    for (int t = 0; t < 4; ++t) {
      const bhalf* bp = wh + (size_t)(t * 16 + l16) * DMODEL + kc * 32 + half * 8;
      acc[t] = wmma_bf16(a, ld_frag16(bp), acc[t]);
    }
  }

#pragma unroll
  for (int t = 0; t < 4; ++t) {
    const int k = t * 16 + l16;
    const float bv_ = bias[h * DKH + k];
#pragma unroll
    for (int r = 0; r < 8; ++r) {
      const int m = mbase + half * 8 + r;     // fused (z,n) row
      const int z = m / NSEQ, n = m % NSEQ;
      const float v = acc[t][r] + bv_;
      if (which == 2)
        vo[((size_t)(z * NHEADS + h) * DKH + k) * NSEQ + n] = (bhalf)v;   // V^T
      else if (which == 1)
        ko[((size_t)(z * NHEADS + h) * NSEQ + n) * DKH + k] = (bhalf)v;
      else
        qo[((size_t)(z * NHEADS + h) * NSEQ + n) * DKH + k] = (bhalf)v;
    }
  }
}

// ---- Flash attention core --------------------------------------------------
// grid (N/128, H, Z); 8 waves/block, each wave owns 16 query rows.
// K/V tiles for each 32-key block are staged into LDS once per block via
// async global->LDS copies (double-buffered), then all 8 waves consume them.
#define KSTR 72   // LDS row stride (bf16) for K tile: 32 rows x 64 dk (+pad)
#define VSTR 40   // LDS row stride (bf16) for V tile: 64 rows x 32 keys (+pad)

__global__ void __launch_bounds__(256)
k_attn(const bhalf* __restrict__ qm, const bhalf* __restrict__ km,
       const bhalf* __restrict__ vt, const int* __restrict__ mask,
       bhalf* __restrict__ ctx) {
  __shared__ __attribute__((aligned(16))) bhalf lds_k[2][32][KSTR];
  __shared__ __attribute__((aligned(16))) bhalf lds_v[2][64][VSTR];
  __shared__ __attribute__((aligned(16))) bhalf lds_p[8][16][32];

  const int tid  = threadIdx.x;
  const int lane = tid & 31;
  const int wave = tid >> 5;
  const int half = lane >> 4;
  const int l16  = lane & 15;
  const int h = blockIdx.y, z = blockIdx.z;
  const int qbase = blockIdx.x * 128 + wave * 16;

  const size_t hoff = (size_t)(z * NHEADS + h);
  const bhalf* Q = qm + hoff * NSEQ * DKH;
  const bhalf* K = km + hoff * NSEQ * DKH;
  const bhalf* V = vt + hoff * DKH * NSEQ;          // [dk][n]
  const int*   mz = mask + z * NSEQ;

  // Cooperative staging: 256 threads x (one 16B K unit + one 16B V unit).
  const int krow = tid >> 3, kch = tid & 7;         // K tile: 32 rows x 8 chunks
  const int vrow = tid >> 2, vch = tid & 3;         // V tile: 64 rows x 4 chunks

  // Q tile 16x64 -> two A fragments, kept resident.
  const v16bf aQ0 = ld_frag16(Q + (size_t)(qbase + l16) * DKH +  0 + half * 8);
  const v16bf aQ1 = ld_frag16(Q + (size_t)(qbase + l16) * DKH + 32 + half * 8);

  const v8f vzero = {0.f, 0.f, 0.f, 0.f, 0.f, 0.f, 0.f, 0.f};
  v8f accO[4];
#pragma unroll
  for (int t = 0; t < 4; ++t) accO[t] = vzero;
  float mrow[8], lrow[8];
#pragma unroll
  for (int r = 0; r < 8; ++r) { mrow[r] = -3.0e38f; lrow[r] = 0.f; }

  const float scale = 0.125f;                        // 1/sqrt(64)

  // Prologue: stage tiles for j=0 into buffer 0.
  async_cp16(K + (size_t)(0 + krow) * DKH + kch * 8, &lds_k[0][krow][kch * 8]);
  async_cp16(V + (size_t)vrow * NSEQ + 0 + vch * 8,  &lds_v[0][vrow][vch * 8]);
  wait_async0();
  __syncthreads();

  for (int j = 0; j < NSEQ; j += 32) {
    const int buf = (j >> 5) & 1;
    // Prefetch next key block into the other buffer while we compute.
    if (j + 32 < NSEQ) {
      async_cp16(K + (size_t)(j + 32 + krow) * DKH + kch * 8,
                 &lds_k[buf ^ 1][krow][kch * 8]);
      async_cp16(V + (size_t)vrow * NSEQ + (j + 32) + vch * 8,
                 &lds_v[buf ^ 1][vrow][vch * 8]);
    }

    // S = Q K^T for 32 keys: two 16x16 tiles, 2 WMMA each (dk = 64).
    v8f s0 = vzero, s1 = vzero;
    {
      const bhalf* b00 = &lds_k[buf][l16][half * 8];            // keys 0..15
      s0 = wmma_bf16(aQ0, ld_frag16(b00), s0);
      s0 = wmma_bf16(aQ1, ld_frag16(b00 + 32), s0);
      const bhalf* b10 = &lds_k[buf][16 + l16][half * 8];       // keys 16..31
      s1 = wmma_bf16(aQ0, ld_frag16(b10), s1);
      s1 = wmma_bf16(aQ1, ld_frag16(b10 + 32), s1);
    }
    const bool v0 = mz[j + l16]      != 0;           // key-validity per column
    const bool v1 = mz[j + 16 + l16] != 0;

    float f0[8], f1[8], nm[8], cf[8];
#pragma unroll
    for (int r = 0; r < 8; ++r) {
      f0[r] = v0 ? s0[r] * scale : -1.0e9f;
      f1[r] = v1 ? s1[r] * scale : -1.0e9f;
    }
    // Row max across the 16 lanes of this half (rows r+8*half).
#pragma unroll
    for (int r = 0; r < 8; ++r) {
      float mx = fmaxf(f0[r], f1[r]);
#pragma unroll
      for (int d = 1; d < 16; d <<= 1) mx = fmaxf(mx, __shfl_xor(mx, d, 32));
      nm[r] = fmaxf(mrow[r], mx);
    }
#pragma unroll
    for (int r = 0; r < 8; ++r) {
      cf[r] = __expf(mrow[r] - nm[r]);
      const float p0 = __expf(f0[r] - nm[r]);
      const float p1 = __expf(f1[r] - nm[r]);
      float ps = p0 + p1;
#pragma unroll
      for (int d = 1; d < 16; d <<= 1) ps += __shfl_xor(ps, d, 32);
      lrow[r] = lrow[r] * cf[r] + ps;
      mrow[r] = nm[r];
      lds_p[wave][half * 8 + r][l16]      = (bhalf)p0;   // C-layout -> LDS
      lds_p[wave][half * 8 + r][16 + l16] = (bhalf)p1;
    }
#pragma unroll
    for (int t = 0; t < 4; ++t)
#pragma unroll
      for (int r = 0; r < 8; ++r) accO[t][r] *= cf[r];

    // Wave-private LDS RAW: make ds stores visible before fragment reload.
    asm volatile("s_wait_dscnt 0" ::: "memory");

    v16bf aP = ld_frag16(&lds_p[wave][l16][half * 8]);  // A-frag 16x32 of P
#pragma unroll
    for (int t = 0; t < 4; ++t) {
      const bhalf* bp = &lds_v[buf][t * 16 + l16][half * 8];
      accO[t] = wmma_bf16(aP, ld_frag16(bp), accO[t]);
    }

    // Next buffer must be fully landed before anyone reads it after barrier.
    wait_async0();
    __syncthreads();
  }

  // Epilogue: 1/l, zero invalid query rows, write ctx as [z,n,h*64+k] bf16.
#pragma unroll
  for (int r = 0; r < 8; ++r) {
    const int q = qbase + half * 8 + r;
    const float inv = (mz[q] != 0) ? 1.0f / fmaxf(lrow[r], 1e-20f) : 0.0f;
#pragma unroll
    for (int t = 0; t < 4; ++t)
      ctx[(size_t)(z * NSEQ + q) * DMODEL + h * DKH + t * 16 + l16] =
          (bhalf)(accO[t][r] * inv);
  }
}

// ---- Output projection: out = ctx @ out_w.T  ([4096x1024] x [1024x1024]) ---
__global__ void __launch_bounds__(256)
k_outproj(const bhalf* __restrict__ cm, const bhalf* __restrict__ ow,
          float* __restrict__ out) {
  const int lane = threadIdx.x & 31;
  const int wave = threadIdx.x >> 5;
  const int half = lane >> 4;
  const int l16  = lane & 15;
  const int mbase = blockIdx.x * 128 + wave * 16;
  const int cbase = blockIdx.y * 64;

  const v8f vzero = {0.f, 0.f, 0.f, 0.f, 0.f, 0.f, 0.f, 0.f};
  v8f acc[4];
#pragma unroll
  for (int t = 0; t < 4; ++t) acc[t] = vzero;

  const bhalf* arow = cm + (size_t)(mbase + l16) * DMODEL + half * 8;
  for (int kc = 0; kc < DMODEL / 32; ++kc) {
    v16bf a = ld_frag16(arow + kc * 32);
#pragma unroll
    for (int t = 0; t < 4; ++t) {
      const bhalf* bp = ow + (size_t)(cbase + t * 16 + l16) * DMODEL + kc * 32 + half * 8;
      acc[t] = wmma_bf16(a, ld_frag16(bp), acc[t]);
    }
  }
#pragma unroll
  for (int t = 0; t < 4; ++t)
#pragma unroll
    for (int r = 0; r < 8; ++r)
      out[(size_t)(mbase + half * 8 + r) * DMODEL + cbase + t * 16 + l16] = acc[t][r];
}

// ---------------------------------------------------------------------------
extern "C" void kernel_launch(void* const* d_in, const int* in_sizes, int n_in,
                              void* d_out, int out_size, void* d_ws, size_t ws_size,
                              hipStream_t stream) {
  const float* x    = (const float*)d_in[0];
  const int*   mask = (const int*)  d_in[1];
  const float* qp   = (const float*)d_in[2];
  const float* kp   = (const float*)d_in[3];
  const float* vp   = (const float*)d_in[4];
  const float* qb   = (const float*)d_in[5];
  const float* kb   = (const float*)d_in[6];
  const float* vb   = (const float*)d_in[7];
  const float* ow   = (const float*)d_in[8];
  float* out = (float*)d_out;

  // Workspace partition (bf16 buffers), ~48 MB total.
  char* p = (char*)d_ws;
  auto take = [&](size_t elems) { bhalf* r = (bhalf*)p; p += ((elems * 2 + 255) & ~(size_t)255); return r; };
  bhalf* xb  = take((size_t)MROWS * DMODEL);                 // x in bf16
  bhalf* wqT = take((size_t)NHEADS * DKH * DMODEL);          // transposed weights
  bhalf* wkT = take((size_t)NHEADS * DKH * DMODEL);
  bhalf* wvT = take((size_t)NHEADS * DKH * DMODEL);
  bhalf* owB = take((size_t)DMODEL * DMODEL);                // out_w [o][d]
  bhalf* qm  = take((size_t)ZB * NHEADS * NSEQ * DKH);       // Q [z,h,n,dk]
  bhalf* km  = take((size_t)ZB * NHEADS * NSEQ * DKH);       // K [z,h,n,dk]
  bhalf* vtm = take((size_t)ZB * NHEADS * DKH * NSEQ);       // V^T [z,h,dk,n]
  bhalf* ctx = take((size_t)MROWS * DMODEL);                 // attention output

  const int nx = MROWS * DMODEL;
  k_cvt<<<(nx + 255) / 256, 256, 0, stream>>>(x, xb, nx);
  const int nw = NHEADS * DMODEL * DKH;
  k_cvt_wT<<<(nw + 255) / 256, 256, 0, stream>>>(qp, wqT);
  k_cvt_wT<<<(nw + 255) / 256, 256, 0, stream>>>(kp, wkT);
  k_cvt_wT<<<(nw + 255) / 256, 256, 0, stream>>>(vp, wvT);
  const int no = DMODEL * DMODEL;
  k_cvt<<<(no + 255) / 256, 256, 0, stream>>>(ow, owB, no);

  k_qkv<<<dim3(MROWS / 128, NHEADS, 3), 256, 0, stream>>>(
      xb, wqT, wkT, wvT, qb, kb, vb, qm, km, vtm);

  k_attn<<<dim3(NSEQ / 128, NHEADS, ZB), 256, 0, stream>>>(qm, km, vtm, mask, ctx);

  k_outproj<<<dim3(MROWS / 128, DMODEL / 64), 256, 0, stream>>>(ctx, owB, out);
}
